// DirectionalGraphSage_38732015076057
// MI455X (gfx1250) — compile-verified, matrix operand order checked by reading, not run on hardware
//
#include <hip/hip_runtime.h>
#include <hip/hip_bf16.h>

typedef __bf16 bf16;
typedef __attribute__((ext_vector_type(16))) __bf16 v16bf;
typedef __attribute__((ext_vector_type(8)))  __bf16 v8bf;
typedef __attribute__((ext_vector_type(8)))  float  v8f;
typedef __attribute__((address_space(3)))    bf16   bf16_lds;

#define NN   10000
#define EE   160000
#define NPAD 10016
#define DIN  256
#define DEIN 128
#define DD   256

static __device__ __forceinline__ bf16 f2bf(float f) {
    unsigned u = __builtin_bit_cast(unsigned, f);
    u += 0x7fffu + ((u >> 16) & 1u);           // round-to-nearest-even
    unsigned short s = (unsigned short)(u >> 16);
    return __builtin_bit_cast(bf16, s);
}

static __device__ __forceinline__ v8f wmma_bf16(v16bf a, v16bf b, v8f c) {
    return __builtin_amdgcn_wmma_f32_16x16x32_bf16(false, a, false, b, (short)0, c, false, false);
}

// async copy of 16 bytes (8 bf16) per lane: global -> LDS, tracked by ASYNCcnt
static __device__ __forceinline__ void async_copy16(bf16* lds_dst, const bf16* gsrc) {
    bf16_lds* lp = (bf16_lds*)lds_dst;                 // addrspacecast -> 32-bit LDS address
    asm volatile("global_load_async_to_lds_b128 %0, %1, off"
                 :: "v"(lp), "v"(gsrc) : "memory");
}

static __device__ __forceinline__ void wait_async0() {
#if __has_builtin(__builtin_amdgcn_s_wait_asynccnt)
    __builtin_amdgcn_s_wait_asynccnt(0);
#else
    asm volatile("s_wait_asynccnt 0x0" ::: "memory");
#endif
}

// A fragment (16x32 bf16) from a row-major tile with row stride `stride` elems.
// ISA layout: lane<16 holds K={kb..kb+7, kb+16..kb+23}, lane>=16 shifted by +8.
static __device__ __forceinline__ v16bf frag_A(const bf16* __restrict__ tile, int stride, int kb, int lane) {
    const int r = lane & 15, hf = lane >> 4;
    const bf16* p = tile + r * stride + kb + hf * 8;
    const v8bf lo = *(const v8bf*)(p);
    const v8bf hi = *(const v8bf*)(p + 16);
    v16bf a;
#pragma unroll
    for (int i = 0; i < 8; ++i) { a[i] = lo[i]; a[i + 8] = hi[i]; }
    return a;
}

// B fragment (32x16 bf16) from pre-transposed weights WT[outCol][K] row-major.
static __device__ __forceinline__ v16bf frag_B(const bf16* __restrict__ WT, int K, int n0, int kb, int lane) {
    const int n = lane & 15, hf = lane >> 4;
    const bf16* p = WT + (size_t)(n0 + n) * K + kb + hf * 8;
    const v8bf lo = *(const v8bf*)(p);
    const v8bf hi = *(const v8bf*)(p + 16);
    v16bf b;
#pragma unroll
    for (int i = 0; i < 8; ++i) { b[i] = lo[i]; b[i + 8] = hi[i]; }
    return b;
}

// ---------------- utility kernels ----------------

__global__ void k_zero(float* __restrict__ p, size_t n) {
    size_t i = (size_t)blockIdx.x * blockDim.x + threadIdx.x;
    size_t st = (size_t)gridDim.x * blockDim.x;
    for (; i < n; i += st) p[i] = 0.0f;
}

// W[R][C] f32 row-major -> WT[C][R] bf16 row-major
__global__ void k_tr(const float* __restrict__ src, bf16* __restrict__ dst, int R, int C) {
    int i = blockIdx.x * blockDim.x + threadIdx.x;
    if (i < R * C) {
        int r = i / C, c = i % C;
        dst[(size_t)c * R + r] = f2bf(src[i]);
    }
}

__global__ void k_deg(const int* __restrict__ eidx, float* __restrict__ deg_in, float* __restrict__ deg_out) {
    int e = blockIdx.x * blockDim.x + threadIdx.x;
    if (e < EE) {
        unsafeAtomicAdd(&deg_out[eidx[e]], 1.0f);       // src -> out-degree
        unsafeAtomicAdd(&deg_in[eidx[EE + e]], 1.0f);   // dst -> in-degree
    }
}

// ---------------- pre_n = relu(x @ Wn + bn), stored bf16 ----------------

__global__ __launch_bounds__(256) void k_pre_n(const float* __restrict__ x,
                                               const bf16* __restrict__ WnT,
                                               const float* __restrict__ bn,
                                               bf16* __restrict__ pre_n) {
    __shared__ alignas(16) bf16 Ax[16 * 256];
    const int t = threadIdx.x, lane = t & 31, w = t >> 5;
    const int r0 = blockIdx.x * 16;
    {   // stage 16 rows of x -> bf16 LDS (clamped rows for padded tail)
        int row = t >> 4, c0 = (t & 15) * 16;
        int grow = r0 + row; if (grow >= NN) grow = NN - 1;
        const float4* s = (const float4*)(x + (size_t)grow * DIN + c0);
        bf16* d = Ax + row * 256 + c0;
#pragma unroll
        for (int q = 0; q < 4; ++q) {
            float4 f = s[q];
            d[q * 4 + 0] = f2bf(f.x); d[q * 4 + 1] = f2bf(f.y);
            d[q * 4 + 2] = f2bf(f.z); d[q * 4 + 3] = f2bf(f.w);
        }
    }
    __syncthreads();
    v8f c0a = {}, c1a = {};
    const int n0a = w * 16, n0b = (w + 8) * 16;
#pragma unroll
    for (int s = 0; s < 8; ++s) {
        const int kb = s * 32;
        v16bf a = frag_A(Ax, 256, kb, lane);
        c0a = wmma_bf16(a, frag_B(WnT, 256, n0a, kb, lane), c0a);
        c1a = wmma_bf16(a, frag_B(WnT, 256, n0b, kb, lane), c1a);
    }
    const int n = lane & 15, hf = lane >> 4;
    const float ba = bn[n0a + n], bb = bn[n0b + n];
#pragma unroll
    for (int v = 0; v < 8; ++v) {
        const int m = hf * 8 + v;
        const size_t grow = (size_t)(r0 + m);
        pre_n[grow * DD + n0a + n] = f2bf(fmaxf(c0a[v] + ba, 0.0f));
        pre_n[grow * DD + n0b + n] = f2bf(fmaxf(c1a[v] + bb, 0.0f));
    }
}

// ---------------- fused per-edge kernel: pre_e, msg_p, msg_s, edge_out ----------------

__global__ __launch_bounds__(256) void k_edge(const float* __restrict__ ef,
                                              const int* __restrict__ eidx,
                                              const bf16* __restrict__ pre_n,
                                              const bf16* __restrict__ WeT,  const float* __restrict__ be,
                                              const bf16* __restrict__ WpaT, const float* __restrict__ bpa,
                                              const bf16* __restrict__ WsaT, const float* __restrict__ bsa,
                                              const bf16* __restrict__ WetT, const float* __restrict__ bet,
                                              float* __restrict__ agg_p, float* __restrict__ agg_s,
                                              float* __restrict__ edge_out) {
    __shared__ alignas(16) bf16 Xe[16 * 128];   // edge_feat tile (bf16)
    __shared__ alignas(16) bf16 An[16 * 256];   // pre_n[src]
    __shared__ alignas(16) bf16 Ad[16 * 256];   // pre_n[dst]
    __shared__ alignas(16) bf16 Pe[16 * 256];   // pre_e tile
    __shared__ int sSrc[16], sDst[16];

    const int t = threadIdx.x, lane = t & 31, w = t >> 5;
    const int eb = blockIdx.x * 16;

    if (t < 16)       sSrc[t]      = eidx[eb + t];
    else if (t < 32)  sDst[t - 16] = eidx[EE + eb + (t - 16)];
    {   // stage edge_feat rows -> bf16 (needs f32->bf16 conversion, VGPR path)
        int row = t >> 4, c0 = (t & 15) * 8;
        const float4* s = (const float4*)(ef + (size_t)(eb + row) * DEIN + c0);
        bf16* d = Xe + row * 128 + c0;
#pragma unroll
        for (int q = 0; q < 2; ++q) {
            float4 f = s[q];
            d[q * 4 + 0] = f2bf(f.x); d[q * 4 + 1] = f2bf(f.y);
            d[q * 4 + 2] = f2bf(f.z); d[q * 4 + 3] = f2bf(f.w);
        }
    }
    __syncthreads();   // sSrc/sDst + Xe visible

    {   // issue async global->LDS gathers for pre_n[src]/pre_n[dst];
        // they complete in the shadow of the pre_e WMMA loop below
        int row = t >> 4, c0 = (t & 15) * 16;
        const bf16* sa = pre_n + (size_t)sSrc[row] * DD + c0;
        const bf16* sd = pre_n + (size_t)sDst[row] * DD + c0;
        async_copy16(An + row * 256 + c0,     sa);
        async_copy16(An + row * 256 + c0 + 8, sa + 8);
        async_copy16(Ad + row * 256 + c0,     sd);
        async_copy16(Ad + row * 256 + c0 + 8, sd + 8);
    }

    const int n0a = w * 16, n0b = (w + 8) * 16;
    const int n = lane & 15, hf = lane >> 4;

    // ---- pre_e = relu(Xe @ We + be) -> Pe (LDS, bf16) ----
    {
        v8f c0a = {}, c1a = {};
#pragma unroll
        for (int s = 0; s < 4; ++s) {
            const int kb = s * 32;
            v16bf a = frag_A(Xe, 128, kb, lane);
            c0a = wmma_bf16(a, frag_B(WeT, 128, n0a, kb, lane), c0a);
            c1a = wmma_bf16(a, frag_B(WeT, 128, n0b, kb, lane), c1a);
        }
        const float ba = be[n0a + n], bb = be[n0b + n];
#pragma unroll
        for (int v = 0; v < 8; ++v) {
            const int m = hf * 8 + v;
            Pe[m * 256 + n0a + n] = f2bf(fmaxf(c0a[v] + ba, 0.0f));
            Pe[m * 256 + n0b + n] = f2bf(fmaxf(c1a[v] + bb, 0.0f));
        }
    }
    wait_async0();     // An/Ad gather landed
    __syncthreads();   // ... and Pe complete, visible to all waves

    // ---- msg_p = relu([An|Pe] @ Wpa + bpa) -> atomic add into agg_p[dst] ----
    {
        v8f c0a = {}, c1a = {};
#pragma unroll
        for (int s = 0; s < 16; ++s) {
            const int kb = s * 32;
            v16bf a = (s < 8) ? frag_A(An, 256, kb, lane) : frag_A(Pe, 256, kb - 256, lane);
            c0a = wmma_bf16(a, frag_B(WpaT, 512, n0a, kb, lane), c0a);
            c1a = wmma_bf16(a, frag_B(WpaT, 512, n0b, kb, lane), c1a);
        }
        const float ba = bpa[n0a + n], bb = bpa[n0b + n];
#pragma unroll
        for (int v = 0; v < 8; ++v) {
            const int m = hf * 8 + v;
            const size_t dn = (size_t)sDst[m];
            unsafeAtomicAdd(&agg_p[dn * DD + n0a + n], fmaxf(c0a[v] + ba, 0.0f));
            unsafeAtomicAdd(&agg_p[dn * DD + n0b + n], fmaxf(c1a[v] + bb, 0.0f));
        }
    }

    // ---- msg_s = relu([Ad|Pe] @ Wsa + bsa) -> atomic add into agg_s[src] ----
    {
        v8f c0a = {}, c1a = {};
#pragma unroll
        for (int s = 0; s < 16; ++s) {
            const int kb = s * 32;
            v16bf a = (s < 8) ? frag_A(Ad, 256, kb, lane) : frag_A(Pe, 256, kb - 256, lane);
            c0a = wmma_bf16(a, frag_B(WsaT, 512, n0a, kb, lane), c0a);
            c1a = wmma_bf16(a, frag_B(WsaT, 512, n0b, kb, lane), c1a);
        }
        const float ba = bsa[n0a + n], bb = bsa[n0b + n];
#pragma unroll
        for (int v = 0; v < 8; ++v) {
            const int m = hf * 8 + v;
            const size_t sn = (size_t)sSrc[m];
            unsafeAtomicAdd(&agg_s[sn * DD + n0a + n], fmaxf(c0a[v] + ba, 0.0f));
            unsafeAtomicAdd(&agg_s[sn * DD + n0b + n], fmaxf(c1a[v] + bb, 0.0f));
        }
    }

    // ---- edge_out = [An|Pe|Ad] @ Wet + bet (no relu) ----
    {
        v8f c0a = {}, c1a = {};
#pragma unroll
        for (int s = 0; s < 24; ++s) {
            const int kb = s * 32;
            v16bf a = (s < 8)  ? frag_A(An, 256, kb, lane)
                    : (s < 16) ? frag_A(Pe, 256, kb - 256, lane)
                               : frag_A(Ad, 256, kb - 512, lane);
            c0a = wmma_bf16(a, frag_B(WetT, 768, n0a, kb, lane), c0a);
            c1a = wmma_bf16(a, frag_B(WetT, 768, n0b, kb, lane), c1a);
        }
        const float ba = bet[n0a + n], bb = bet[n0b + n];
#pragma unroll
        for (int v = 0; v < 8; ++v) {
            const int m = hf * 8 + v;
            const size_t row = (size_t)(eb + m);
            edge_out[row * DD + n0a + n] = c0a[v] + ba;
            edge_out[row * DD + n0b + n] = c1a[v] + bb;
        }
    }
}

// ---------------- node_out = [agg_p/deg | pre_n | agg_s/deg] @ Wnt + bnt ----------------

__global__ __launch_bounds__(256) void k_node(const bf16* __restrict__ pre_n,
                                              const float* __restrict__ agg_p,
                                              const float* __restrict__ agg_s,
                                              const float* __restrict__ deg_in,
                                              const float* __restrict__ deg_out,
                                              const bf16* __restrict__ WntT,
                                              const float* __restrict__ bnt,
                                              float* __restrict__ node_out) {
    __shared__ alignas(16) bf16 Tp[16 * 256];
    __shared__ alignas(16) bf16 Tn[16 * 256];
    __shared__ alignas(16) bf16 Ts[16 * 256];
    const int t = threadIdx.x, lane = t & 31, w = t >> 5;
    const int r0 = blockIdx.x * 16;
    {
        int row = t >> 4, c0 = (t & 15) * 16;
        const size_t grow = (size_t)(r0 + row);      // < NPAD; padded rows are zero
        // pre_n rows are already bf16: async copy straight into LDS
        const bf16* pn = pre_n + grow * DD + c0;
        async_copy16(Tn + row * 256 + c0,     pn);
        async_copy16(Tn + row * 256 + c0 + 8, pn + 8);
        const float ri = 1.0f / fmaxf(deg_in[grow], 1.0f);
        const float ro = 1.0f / fmaxf(deg_out[grow], 1.0f);
        const float4* ap = (const float4*)(agg_p + grow * DD + c0);
        const float4* as = (const float4*)(agg_s + grow * DD + c0);
#pragma unroll
        for (int q = 0; q < 4; ++q) {
            float4 fp = ap[q], fs = as[q];
            bf16* dp = Tp + row * 256 + c0 + q * 4;
            bf16* ds = Ts + row * 256 + c0 + q * 4;
            dp[0] = f2bf(fp.x * ri); dp[1] = f2bf(fp.y * ri); dp[2] = f2bf(fp.z * ri); dp[3] = f2bf(fp.w * ri);
            ds[0] = f2bf(fs.x * ro); ds[1] = f2bf(fs.y * ro); ds[2] = f2bf(fs.z * ro); ds[3] = f2bf(fs.w * ro);
        }
    }
    wait_async0();
    __syncthreads();
    v8f c0a = {}, c1a = {};
    const int n0a = w * 16, n0b = (w + 8) * 16;
#pragma unroll
    for (int s = 0; s < 24; ++s) {
        const int kb = s * 32;
        const bf16* tile = (s < 8) ? Tp : (s < 16) ? Tn : Ts;
        v16bf a = frag_A(tile, 256, kb & 255, lane);
        c0a = wmma_bf16(a, frag_B(WntT, 768, n0a, kb, lane), c0a);
        c1a = wmma_bf16(a, frag_B(WntT, 768, n0b, kb, lane), c1a);
    }
    const int n = lane & 15, hf = lane >> 4;
    const float ba = bnt[n0a + n], bb = bnt[n0b + n];
#pragma unroll
    for (int v = 0; v < 8; ++v) {
        const int m = hf * 8 + v;
        const int grow = r0 + m;
        if (grow < NN) {
            node_out[(size_t)grow * DD + n0a + n] = c0a[v] + ba;
            node_out[(size_t)grow * DD + n0b + n] = c1a[v] + bb;
        }
    }
}

// ---------------- host launcher ----------------

extern "C" void kernel_launch(void* const* d_in, const int* in_sizes, int n_in,
                              void* d_out, int out_size, void* d_ws, size_t ws_size,
                              hipStream_t stream) {
    const float* x   = (const float*)d_in[0];
    const float* ef  = (const float*)d_in[1];
    const int*   eix = (const int*)d_in[2];
    const float* Wn  = (const float*)d_in[3];  const float* bn  = (const float*)d_in[4];
    const float* We  = (const float*)d_in[5];  const float* be  = (const float*)d_in[6];
    const float* Wpa = (const float*)d_in[7];  const float* bpa = (const float*)d_in[8];
    const float* Wsa = (const float*)d_in[9];  const float* bsa = (const float*)d_in[10];
    const float* Wnt = (const float*)d_in[11]; const float* bnt = (const float*)d_in[12];
    const float* Wet = (const float*)d_in[13]; const float* bet = (const float*)d_in[14];

    char* ws = (char*)d_ws;
    size_t off = 0;
    bf16*  pre_n   = (bf16*)(ws + off);  off += (size_t)NPAD * DD * 2;   // 5.1 MB
    float* agg_p   = (float*)(ws + off); off += (size_t)NPAD * DD * 4;   // 10.3 MB
    float* agg_s   = (float*)(ws + off); off += (size_t)NPAD * DD * 4;   // 10.3 MB
    float* deg_in  = (float*)(ws + off); off += 40960;
    float* deg_out = (float*)(ws + off); off += 40960;
    bf16*  WnT  = (bf16*)(ws + off); off += (size_t)256 * 256 * 2;
    bf16*  WeT  = (bf16*)(ws + off); off += (size_t)256 * 128 * 2;
    bf16*  WpaT = (bf16*)(ws + off); off += (size_t)256 * 512 * 2;
    bf16*  WsaT = (bf16*)(ws + off); off += (size_t)256 * 512 * 2;
    bf16*  WntT = (bf16*)(ws + off); off += (size_t)256 * 768 * 2;
    bf16*  WetT = (bf16*)(ws + off); off += (size_t)256 * 768 * 2;

    float* node_out = (float*)d_out;
    float* edge_out = node_out + (size_t)NN * DD;

    // zero accumulators + degrees (agg_p..deg_out are contiguous)
    const size_t nzero = 2 * (size_t)NPAD * DD + 2 * 10240;
    k_zero<<<2048, 256, 0, stream>>>(agg_p, nzero);

    // transpose+convert weights to bf16 [outCol][K]
    k_tr<<<(256 * 256 + 255) / 256, 256, 0, stream>>>(Wn,  WnT,  256, 256);
    k_tr<<<(128 * 256 + 255) / 256, 256, 0, stream>>>(We,  WeT,  128, 256);
    k_tr<<<(512 * 256 + 255) / 256, 256, 0, stream>>>(Wpa, WpaT, 512, 256);
    k_tr<<<(512 * 256 + 255) / 256, 256, 0, stream>>>(Wsa, WsaT, 512, 256);
    k_tr<<<(768 * 256 + 255) / 256, 256, 0, stream>>>(Wnt, WntT, 768, 256);
    k_tr<<<(768 * 256 + 255) / 256, 256, 0, stream>>>(Wet, WetT, 768, 256);

    k_pre_n<<<NPAD / 16, 256, 0, stream>>>(x, WnT, bn, pre_n);
    k_deg<<<(EE + 255) / 256, 256, 0, stream>>>(eix, deg_in, deg_out);

    k_edge<<<EE / 16, 256, 0, stream>>>(ef, eix, pre_n,
                                        WeT, be, WpaT, bpa, WsaT, bsa, WetT, bet,
                                        agg_p, agg_s, edge_out);

    k_node<<<NPAD / 16, 256, 0, stream>>>(pre_n, agg_p, agg_s, deg_in, deg_out,
                                          WntT, bnt, node_out);
}
